// AppUsageFEDformer_86268713108110
// MI455X (gfx1250) — compile-verified, matrix operand order checked by reading, not run on hardware
//
#include <hip/hip_runtime.h>
#include <hip/hip_bf16.h>

typedef __bf16 bf16;
typedef __attribute__((ext_vector_type(16))) __bf16 v16bf;
typedef __attribute__((ext_vector_type(8)))  __bf16 v8bf;
typedef __attribute__((ext_vector_type(4)))  __bf16 v4bf;
typedef __attribute__((ext_vector_type(8)))  float  v8f;
typedef __attribute__((ext_vector_type(4)))  float  fv4;

#define Bsz   64
#define Lsz   512
#define Dsz   512
#define Hsz   8
#define Esz   64
#define DFFsz 2048
#define MODES 32
#define NAPP  10000
#define KPROJ 536   // D + 24

// ---------------------------------------------------------------------------
// Tiled bf16 WMMA GEMM:  C[M,N] = A[M,K] * W[N,K]^T (+bias) (optional ReLU)
// Block: 256 threads = 8 waves; tile 128(M) x 128(N), BK=32.
// Wave grid 4(M) x 2(N); each wave -> 32x64 = 2x4 wmma 16x16 tiles.
// FULL path: register-staged, LDS double-buffered, one barrier per K-step.
// ---------------------------------------------------------------------------
__device__ __forceinline__
void gemm_full(const float* __restrict__ A, const float* __restrict__ W,
               const float* __restrict__ bias, float* __restrict__ C,
               int M, int N, int K, int act,
               bf16 (*sA)[128][40], bf16 (*sB)[128][40]) {
    constexpr int BM = 128, BN = 128, BK = 32;

    const int tid   = threadIdx.x;
    const int lane  = tid & 31;
    const int wave  = tid >> 5;
    const int wm    = wave & 3;
    const int wn    = wave >> 2;
    const int row16 = lane & 15;
    const int hi    = lane >> 4;
    const int kb    = hi * 8;

    const int blockM = blockIdx.y * BM;
    const int blockN = blockIdx.x * BN;

    const int r0 = tid >> 3;           // 0..31
    const int cg = (tid & 7) * 4;      // 0,4,...,28

    // Per-thread staging pointers, advanced by BK each step.
    const float* pa[4];
    const float* pw[4];
    #pragma unroll
    for (int i = 0; i < 4; ++i) {
        pa[i] = A + (size_t)(blockM + r0 + 32 * i) * K + cg;
        pw[i] = W + (size_t)(blockN + r0 + 32 * i) * K + cg;
    }

    v8f acc[2][4] = {};
    fv4 ra[4], rw[4];

    // Prologue: stage tile 0 into registers.
    #pragma unroll
    for (int i = 0; i < 4; ++i) {
        ra[i] = *(const fv4*)pa[i];
        rw[i] = *(const fv4*)pw[i];
        pa[i] += BK; pw[i] += BK;
    }

    int buf = 0;
    for (int k0 = 0; k0 < K; k0 += BK) {
        // ---- commit staged registers to LDS[buf] ----
        #pragma unroll
        for (int i = 0; i < 4; ++i) {
            *(v4bf*)&sA[buf][r0 + 32 * i][cg] = __builtin_convertvector(ra[i], v4bf);
            *(v4bf*)&sB[buf][r0 + 32 * i][cg] = __builtin_convertvector(rw[i], v4bf);
        }
        __syncthreads();

        // ---- stage next tile (overlaps with WMMA below) ----
        if (k0 + BK < K) {
            #pragma unroll
            for (int i = 0; i < 4; ++i) {
                ra[i] = *(const fv4*)pa[i];
                rw[i] = *(const fv4*)pw[i];
                pa[i] += BK; pw[i] += BK;
            }
            if (k0 + 2 * BK < K) {
                #pragma unroll
                for (int i = 0; i < 4; ++i) {
                    __builtin_prefetch(pa[i], 0, 1);
                    __builtin_prefetch(pw[i], 0, 1);
                }
            }
        }

        // ---- fragments per documented 16-bit A/B layouts ----
        v16bf afr[2], bfr[4];
        #pragma unroll
        for (int ti = 0; ti < 2; ++ti) {
            int ar = wm * 32 + ti * 16 + row16;
            v8bf lo = *(const v8bf*)&sA[buf][ar][kb];
            v8bf h8 = *(const v8bf*)&sA[buf][ar][kb + 16];
            afr[ti] = __builtin_shufflevector(lo, h8, 0,1,2,3,4,5,6,7,8,9,10,11,12,13,14,15);
        }
        #pragma unroll
        for (int tj = 0; tj < 4; ++tj) {
            int br = wn * 64 + tj * 16 + row16;
            v8bf lo = *(const v8bf*)&sB[buf][br][kb];
            v8bf h8 = *(const v8bf*)&sB[buf][br][kb + 16];
            bfr[tj] = __builtin_shufflevector(lo, h8, 0,1,2,3,4,5,6,7,8,9,10,11,12,13,14,15);
        }
        #pragma unroll
        for (int ti = 0; ti < 2; ++ti)
            #pragma unroll
            for (int tj = 0; tj < 4; ++tj)
                acc[ti][tj] = __builtin_amdgcn_wmma_f32_16x16x32_bf16(
                    false, afr[ti], false, bfr[tj], (short)0, acc[ti][tj], false, false);

        buf ^= 1;
    }

    // ---- epilogue (interior block: no guards) ----
    #pragma unroll
    for (int ti = 0; ti < 2; ++ti) {
        #pragma unroll
        for (int tj = 0; tj < 4; ++tj) {
            int n = blockN + wn * 64 + tj * 16 + row16;
            float bv = bias ? bias[n] : 0.f;
            #pragma unroll
            for (int v = 0; v < 8; ++v) {
                int m = blockM + wm * 32 + ti * 16 + v + hi * 8;
                float val = acc[ti][tj][v] + bv;
                if (act) val = fmaxf(val, 0.f);
                C[(size_t)m * N + n] = val;
            }
        }
    }
}

// Guarded edge path (only used by the small projection GEMM / ragged blocks).
__device__ __forceinline__
void gemm_edge(const float* __restrict__ A, const float* __restrict__ W,
               const float* __restrict__ bias, float* __restrict__ C,
               int M, int N, int K, int act,
               bf16 (*sA)[128][40], bf16 (*sB)[128][40]) {
    constexpr int BM = 128, BN = 128, BK = 32;

    const int tid   = threadIdx.x;
    const int lane  = tid & 31;
    const int wave  = tid >> 5;
    const int wm    = wave & 3;
    const int wn    = wave >> 2;
    const int row16 = lane & 15;
    const int hi    = lane >> 4;
    const int kb    = hi * 8;

    const int blockM = blockIdx.y * BM;
    const int blockN = blockIdx.x * BN;

    v8f acc[2][4] = {};

    const int r0 = tid >> 3;
    const int cg = (tid & 7) * 4;

    for (int k0 = 0; k0 < K; k0 += BK) {
        const int kk = k0 + cg;
        #pragma unroll
        for (int i = 0; i < 4; ++i) {
            int row = r0 + 32 * i;
            int gm  = blockM + row;
            fv4 v = {0.f, 0.f, 0.f, 0.f};
            if (gm < M && kk < K) {
                const float* p = A + (size_t)gm * K;
                if (kk + 4 <= K) v = *(const fv4*)(p + kk);
                else {
                    v.x = p[kk];
                    if (kk + 1 < K) v.y = p[kk + 1];
                    if (kk + 2 < K) v.z = p[kk + 2];
                    if (kk + 3 < K) v.w = p[kk + 3];
                }
            }
            *(v4bf*)&sA[0][row][cg] = __builtin_convertvector(v, v4bf);
        }
        #pragma unroll
        for (int i = 0; i < 4; ++i) {
            int row = r0 + 32 * i;
            int gn  = blockN + row;
            fv4 v = {0.f, 0.f, 0.f, 0.f};
            if (gn < N && kk < K) {
                const float* p = W + (size_t)gn * K;
                if (kk + 4 <= K) v = *(const fv4*)(p + kk);
                else {
                    v.x = p[kk];
                    if (kk + 1 < K) v.y = p[kk + 1];
                    if (kk + 2 < K) v.z = p[kk + 2];
                    if (kk + 3 < K) v.w = p[kk + 3];
                }
            }
            *(v4bf*)&sB[0][row][cg] = __builtin_convertvector(v, v4bf);
        }
        __syncthreads();

        v16bf afr[2], bfr[4];
        #pragma unroll
        for (int ti = 0; ti < 2; ++ti) {
            int ar = wm * 32 + ti * 16 + row16;
            v8bf lo = *(const v8bf*)&sA[0][ar][kb];
            v8bf h8 = *(const v8bf*)&sA[0][ar][kb + 16];
            afr[ti] = __builtin_shufflevector(lo, h8, 0,1,2,3,4,5,6,7,8,9,10,11,12,13,14,15);
        }
        #pragma unroll
        for (int tj = 0; tj < 4; ++tj) {
            int br = wn * 64 + tj * 16 + row16;
            v8bf lo = *(const v8bf*)&sB[0][br][kb];
            v8bf h8 = *(const v8bf*)&sB[0][br][kb + 16];
            bfr[tj] = __builtin_shufflevector(lo, h8, 0,1,2,3,4,5,6,7,8,9,10,11,12,13,14,15);
        }
        #pragma unroll
        for (int ti = 0; ti < 2; ++ti)
            #pragma unroll
            for (int tj = 0; tj < 4; ++tj)
                acc[ti][tj] = __builtin_amdgcn_wmma_f32_16x16x32_bf16(
                    false, afr[ti], false, bfr[tj], (short)0, acc[ti][tj], false, false);
        __syncthreads();
    }

    #pragma unroll
    for (int ti = 0; ti < 2; ++ti) {
        #pragma unroll
        for (int tj = 0; tj < 4; ++tj) {
            int n = blockN + wn * 64 + tj * 16 + row16;
            if (n >= N) continue;
            float bv = bias ? bias[n] : 0.f;
            #pragma unroll
            for (int v = 0; v < 8; ++v) {
                int m = blockM + wm * 32 + ti * 16 + v + hi * 8;
                if (m < M) {
                    float val = acc[ti][tj][v] + bv;
                    if (act) val = fmaxf(val, 0.f);
                    C[(size_t)m * N + n] = val;
                }
            }
        }
    }
}

__global__ __launch_bounds__(256)
void gemm_bf16_wmma(const float* __restrict__ A, const float* __restrict__ W,
                    const float* __restrict__ bias, float* __restrict__ C,
                    int M, int N, int K, int act) {
    __shared__ __align__(16) bf16 sA[2][128][40];
    __shared__ __align__(16) bf16 sB[2][128][40];
    const bool full = ((int)blockIdx.y * 128 + 128 <= M) &&
                      ((int)blockIdx.x * 128 + 128 <= N) &&
                      ((K & 31) == 0);
    if (full) gemm_full(A, W, bias, C, M, N, K, act, sA, sB);
    else      gemm_edge(A, W, bias, C, M, N, K, act, sA, sB);
}

// ---------------------------------------------------------------------------
// Embedding:  X[b,l,d] = emb[x_app[b,l], d] + x_time[b,l]*tw[d] + tb[d]
// ---------------------------------------------------------------------------
__global__ void embed_kernel(const int* __restrict__ x_app, const float* __restrict__ x_time,
                             const float* __restrict__ emb, const float* __restrict__ tw,
                             const float* __restrict__ tb, float* __restrict__ X, int total) {
    int idx = blockIdx.x * 256 + threadIdx.x;
    if (idx >= total) return;
    int d  = idx & (Dsz - 1);
    int bl = idx >> 9;
    int a  = x_app[bl];
    X[idx] = emb[(size_t)a * Dsz + d] + x_time[bl] * tw[d] + tb[d];
}

__global__ void init_tables(float* __restrict__ cosT, float* __restrict__ sinT) {
    int idx = blockIdx.x * 256 + threadIdx.x;
    if (idx >= MODES * Lsz) return;
    int m = idx >> 9, l = idx & (Lsz - 1);
    float ang = 6.283185307179586f * (float)(m * l) / (float)Lsz;
    cosT[idx] = cosf(ang);
    sinT[idx] = sinf(ang);
}

// Forward DFT (first 32 modes) over the L axis of q[b,l,h,e]; block = (b,h,e)
__global__ void dft_fwd(const float* __restrict__ q, const float* __restrict__ cosT,
                        const float* __restrict__ sinT, float* __restrict__ Xre,
                        float* __restrict__ Xim) {
    __shared__ float sQ[Lsz];
    __shared__ float sP[256];
    int bid = blockIdx.x;                       // ((b*H+h)*E + e)
    int e = bid & 63, h = (bid >> 6) & 7, b = bid >> 9;
    const float* src = q + ((size_t)b * Lsz) * Dsz + h * Esz + e;
    int t = threadIdx.x;
    sQ[t]       = src[(size_t)t * Dsz];
    sQ[t + 256] = src[(size_t)(t + 256) * Dsz];
    __syncthreads();
    int m = t & 31, reim = (t >> 5) & 1, p = t >> 6;
    const float* tab = (reim == 0) ? cosT : sinT;
    float s = 0.f;
    int l0 = p * 128;
    for (int l = l0; l < l0 + 128; ++l) s += sQ[l] * tab[m * Lsz + l];
    sP[t] = s;
    __syncthreads();
    if (t < 64) {
        float tot = sP[t] + sP[t + 64] + sP[t + 128] + sP[t + 192];
        size_t base = (size_t)bid * MODES + m;  // layout (b,h,e,m)
        if (reim == 0) Xre[base] = tot; else Xim[base] = -tot;
    }
}

// Complex mode mix: Y[b,h,o,m] = sum_i X[b,h,i,m] * (wr + i*wi)[h,i,o,m]; block = (b,h)
__global__ void mode_mix(const float* __restrict__ Xre, const float* __restrict__ Xim,
                         const float* __restrict__ wr, const float* __restrict__ wi,
                         float* __restrict__ Yre, float* __restrict__ Yim) {
    __shared__ float sXr[Esz * MODES];
    __shared__ float sXi[Esz * MODES];
    int bid = blockIdx.x;                       // b*H + h
    int h = bid & 7;
    size_t base = (size_t)bid * (Esz * MODES);
    int t = threadIdx.x;
    #pragma unroll
    for (int j = 0; j < 8; ++j) {
        int k = t + j * 256;
        sXr[k] = Xre[base + k];
        sXi[k] = Xim[base + k];
    }
    __syncthreads();
    size_t wbase = (size_t)h * Esz * Esz * MODES;
    #pragma unroll
    for (int j = 0; j < 8; ++j) {
        int oi = t + j * 256;
        int o = oi >> 5, m = oi & 31;
        float ar = 0.f, ai = 0.f;
        for (int i = 0; i < Esz; ++i) {
            float xr = sXr[i * MODES + m], xi = sXi[i * MODES + m];
            size_t widx = wbase + ((size_t)i * Esz + o) * MODES + m;
            float wrv = wr[widx], wiv = wi[widx];
            ar += xr * wrv - xi * wiv;
            ai += xr * wiv + xi * wrv;
        }
        Yre[base + oi] = ar;                    // layout (b,h,o,m)
        Yim[base + oi] = ai;
    }
}

// Inverse rfft with only 32 modes; block = (b,h,o).
// NOTE: writes [b, channel, l] to replicate reference's raw reshape of (B,H,E,L)
// into (B,L,H*E) (valid only because H*E == L == 512).
__global__ void dft_inv(const float* __restrict__ Yre, const float* __restrict__ Yim,
                        const float* __restrict__ cosT, const float* __restrict__ sinT,
                        float* __restrict__ out) {
    __shared__ float sYr[MODES];
    __shared__ float sYi[MODES];
    int bid = blockIdx.x;                       // ((b*H+h)*E + o)
    int o = bid & 63, h = (bid >> 6) & 7, b = bid >> 9;
    int t = threadIdx.x;
    if (t < MODES) {
        sYr[t] = Yre[(size_t)bid * MODES + t];
        sYi[t] = Yim[(size_t)bid * MODES + t];
    }
    __syncthreads();
    float y0 = sYr[0];
    #pragma unroll
    for (int j = 0; j < 2; ++j) {
        int l = t + j * 256;
        float acc = y0;
        for (int m = 1; m < MODES; ++m)
            acc += 2.f * (sYr[m] * cosT[m * Lsz + l] - sYi[m] * sinT[m * Lsz + l]);
        out[(((size_t)b * Lsz) + (h * Esz + o)) * Dsz + l] = acc * (1.f / (float)Lsz);
    }
}

// Fused residual add + series_decomp (25-tap edge-padded moving average):
// out = (a+b2) - movavg(a+b2), along axis 1 of (B,512,512)
__global__ void add_decomp(const float* __restrict__ a, const float* __restrict__ b2,
                           float* __restrict__ out, int total) {
    int idx = blockIdx.x * 256 + threadIdx.x;
    if (idx >= total) return;
    int d = idx & (Dsz - 1);
    int l = (idx >> 9) & (Lsz - 1);
    int b = idx >> 18;
    const float* ab = a  + (size_t)b * (Lsz * Dsz) + d;
    const float* bb = b2 + (size_t)b * (Lsz * Dsz) + d;
    float s = 0.f;
    #pragma unroll
    for (int j = -12; j <= 12; ++j) {
        int jc = l + j;
        jc = jc < 0 ? 0 : (jc > Lsz - 1 ? Lsz - 1 : jc);
        s += ab[(size_t)jc * Dsz] + bb[(size_t)jc * Dsz];
    }
    float tv = ab[(size_t)l * Dsz] + bb[(size_t)l * Dsz];
    out[idx] = tv - s * (1.f / 25.f);
}

// Row layernorm over last axis (D=512); block = one row of (B*L)
__global__ void ln_rows(const float* __restrict__ x, const float* __restrict__ w,
                        const float* __restrict__ bnb, float* __restrict__ out) {
    __shared__ float red[256];
    size_t row = blockIdx.x;
    const float* xr = x + row * Dsz;
    int t = threadIdx.x;
    float v0 = xr[t], v1 = xr[t + 256];
    red[t] = v0 + v1;
    __syncthreads();
    for (int s = 128; s > 0; s >>= 1) { if (t < s) red[t] += red[t + s]; __syncthreads(); }
    float mu = red[0] * (1.f / (float)Dsz);
    __syncthreads();
    float d0 = v0 - mu, d1 = v1 - mu;
    red[t] = d0 * d0 + d1 * d1;
    __syncthreads();
    for (int s = 128; s > 0; s >>= 1) { if (t < s) red[t] += red[t + s]; __syncthreads(); }
    float var = red[0] * (1.f / (float)Dsz);
    float rs  = rsqrtf(var + 1e-5f);
    out[row * Dsz + t]       = d0 * rs * w[t] + bnb[t];
    out[row * Dsz + t + 256] = d1 * rs * w[t + 256] + bnb[t + 256];
}

// Subtract per-(b,d) mean over L, keep last token, concat time_vecs[:, -1, :]
__global__ void ln_cols_concat(const float* __restrict__ xh, const float* __restrict__ time_vecs,
                               float* __restrict__ P) {
    int b = blockIdx.x;
    int t = threadIdx.x;
    #pragma unroll
    for (int j = 0; j < 2; ++j) {
        int d = t + j * 256;
        const float* col = xh + ((size_t)b * Lsz) * Dsz + d;
        float s = 0.f;
        for (int l = 0; l < Lsz; ++l) s += col[(size_t)l * Dsz];
        P[(size_t)b * KPROJ + d] = col[(size_t)(Lsz - 1) * Dsz] - s * (1.f / (float)Lsz);
    }
    if (t < 24)
        P[(size_t)b * KPROJ + Dsz + t] = time_vecs[((size_t)b * Lsz + (Lsz - 1)) * 24 + t];
}

// ---------------------------------------------------------------------------
extern "C" void kernel_launch(void* const* d_in, const int* in_sizes, int n_in,
                              void* d_out, int out_size, void* d_ws, size_t ws_size,
                              hipStream_t stream) {
    const int*   x_app    = (const int*)  d_in[0];
    const float* x_time   = (const float*)d_in[1];
    const float* time_vecs= (const float*)d_in[2];
    // d_in[3] targets: unused
    const float* emb      = (const float*)d_in[4];
    const float* time_w   = (const float*)d_in[5];
    const float* time_b   = (const float*)d_in[6];
    const float* Wq       = (const float*)d_in[7];
    const float* bq       = (const float*)d_in[8];
    const float* Wo       = (const float*)d_in[9];
    const float* bo       = (const float*)d_in[10];
    const float* four_wr  = (const float*)d_in[11];
    const float* four_wi  = (const float*)d_in[12];
    const float* conv1    = (const float*)d_in[13];
    const float* conv2    = (const float*)d_in[14];
    const float* norm_w   = (const float*)d_in[15];
    const float* norm_b   = (const float*)d_in[16];
    const float* proj_w   = (const float*)d_in[17];
    const float* proj_b   = (const float*)d_in[18];
    float* out = (float*)d_out;

    const int BL = Bsz * Lsz;                 // 32768
    const size_t NE   = (size_t)BL * Dsz;     // 16,777,216
    const int MCH = 4096;                     // FFN row chunk
    const size_t YSZ  = (size_t)MCH * DFFsz;  // 8,388,608
    const size_t SPEC = (size_t)Bsz * Hsz * Esz * MODES; // 1,048,576
    const size_t TAB  = (size_t)MODES * Lsz;  // 16,384

    float* ws   = (float*)d_ws;
    float* X    = ws;
    float* T1   = X   + NE;
    float* T2   = T1  + NE;
    float* Y    = T2  + NE;
    float* Xre  = Y   + YSZ;
    float* Xim  = Xre + SPEC;
    float* Yre  = Xim + SPEC;
    float* Yim  = Yre + SPEC;
    float* cosT = Yim + SPEC;
    float* sinT = cosT + TAB;
    float* P    = sinT + TAB;

    auto gemm = [&](const float* A, const float* W, const float* bias, float* Cp,
                    int M, int N, int K, int act) {
        dim3 grid((N + 127) / 128, (M + 127) / 128);
        gemm_bf16_wmma<<<grid, 256, 0, stream>>>(A, W, bias, Cp, M, N, K, act);
    };

    init_tables<<<(MODES * Lsz + 255) / 256, 256, 0, stream>>>(cosT, sinT);

    int total = (int)NE;
    embed_kernel<<<(total + 255) / 256, 256, 0, stream>>>(x_app, x_time, emb, time_w, time_b,
                                                          X, total);

    for (int l = 0; l < 2; ++l) {
        const float* Wq_l = Wq + (size_t)l * Dsz * Dsz;
        const float* bq_l = bq + (size_t)l * Dsz;
        const float* Wo_l = Wo + (size_t)l * Dsz * Dsz;
        const float* bo_l = bo + (size_t)l * Dsz;
        const float* wr_l = four_wr + (size_t)l * Hsz * Esz * Esz * MODES;
        const float* wi_l = four_wi + (size_t)l * Hsz * Esz * Esz * MODES;
        const float* c1_l = conv1 + (size_t)l * DFFsz * Dsz;
        const float* c2_l = conv2 + (size_t)l * Dsz * DFFsz;

        // q = X @ Wq^T + bq
        gemm(X, Wq_l, bq_l, T1, BL, Dsz, Dsz, 0);
        // spectral block
        dft_fwd<<<Bsz * Hsz * Esz, 256, 0, stream>>>(T1, cosT, sinT, Xre, Xim);
        mode_mix<<<Bsz * Hsz, 256, 0, stream>>>(Xre, Xim, wr_l, wi_l, Yre, Yim);
        dft_inv<<<Bsz * Hsz * Esz, 256, 0, stream>>>(Yre, Yim, cosT, sinT, T2);
        // new_x = fourier_out @ Wo^T + bo
        gemm(T2, Wo_l, bo_l, T1, BL, Dsz, Dsz, 0);
        // x = decomp(X + new_x)  -> T2
        add_decomp<<<(total + 255) / 256, 256, 0, stream>>>(X, T1, T2, total);
        // FFN (chunked over rows): y = relu(x @ c1^T) @ c2^T -> T1
        for (int c = 0; c < BL / MCH; ++c) {
            const float* Ac = T2 + (size_t)c * MCH * Dsz;
            float* Oc = T1 + (size_t)c * MCH * Dsz;
            gemm(Ac, c1_l, nullptr, Y, MCH, DFFsz, Dsz, 1);
            gemm(Y, c2_l, nullptr, Oc, MCH, Dsz, DFFsz, 0);
        }
        // x = decomp(x + y) -> X
        add_decomp<<<(total + 255) / 256, 256, 0, stream>>>(T2, T1, X, total);
    }

    // layernorm + column-mean subtract + concat + projection
    ln_rows<<<BL, 256, 0, stream>>>(X, norm_w, norm_b, T1);
    ln_cols_concat<<<Bsz, 256, 0, stream>>>(T1, time_vecs, P);
    gemm(P, proj_w, proj_b, out, Bsz, NAPP, KPROJ, 0);

    (void)in_sizes; (void)n_in; (void)out_size; (void)ws_size;
}
